// MultiHeadAttentionClassical_65481071401252
// MI455X (gfx1250) — compile-verified, matrix operand order checked by reading, not run on hardware
//
#include <hip/hip_runtime.h>
#include <hip/hip_bf16.h>
#include <stdint.h>

typedef __attribute__((ext_vector_type(16))) _Float16 v16h;
typedef __attribute__((ext_vector_type(8)))  _Float16 v8h;
typedef __attribute__((ext_vector_type(4)))  _Float16 v4h;
typedef __attribute__((ext_vector_type(8)))  float    v8f;

#define B_  2
#define S_  2048
#define E_  1024
#define H_  16
#define DK_ 64
#define M_  (B_*S_)   // 4096 rows total

union FragU { v16h v; struct { v8h lo, hi; } p; };

// A/B fragment for v_wmma_f32_16x16x32_f16 from a contiguous f16 "row".
// Element mapping (ISA 7.12.2): e in [0,8): K = half*8 + e ; e in [8,16): K = 16 + half*8 + (e-8)
__device__ __forceinline__ v16h load_frag(const _Float16* rowBase, int half) {
  FragU u;
  u.p.lo = *(const v8h*)(rowBase + half * 8);
  u.p.hi = *(const v8h*)(rowBase + 16 + half * 8);
  return u.v;
}

__device__ __forceinline__ v8f wmma_f16(v16h a, v16h b, v8f c) {
  return __builtin_amdgcn_wmma_f32_16x16x32_f16(false, a, false, b, (short)0, c, false, false);
}

// Async global->LDS 16B copy (CDNA5, ASYNCcnt-tracked). LDS dest offset is the
// low 32 bits of the flat shared-memory address (LDS aperture: {base, off[31:0]}).
__device__ __forceinline__ void async_ld16(const _Float16* g, _Float16* l) {
  uint32_t loff = (uint32_t)(uintptr_t)l;
  asm volatile("global_load_async_to_lds_b128 %0, %1, off"
               :: "v"(loff), "v"(g) : "memory");
}

// ---------------------------------------------------------------- convert
__global__ __launch_bounds__(256) void cvt_f32_f16(const float* __restrict__ src,
                                                   _Float16* __restrict__ dst, int n4) {
  int i = blockIdx.x * blockDim.x + threadIdx.x;
  int stride = gridDim.x * blockDim.x;
  for (; i < n4; i += stride) {
    float4 f = ((const float4*)src)[i];
    v4h h = { (_Float16)f.x, (_Float16)f.y, (_Float16)f.z, (_Float16)f.w };
    ((v4h*)dst)[i] = h;
  }
}

// ---------------------------------------------------------------- fused QKV projection
// y = x @ W^T + b for Wq/Wk/Wv. One wave = one 16x16 tile; software-pipelined
// so the next K-chunk's fragments are in flight while the current WMMAs run.
__global__ __launch_bounds__(256) void qkv_proj(
    const _Float16* __restrict__ xh,
    const _Float16* __restrict__ wq, const _Float16* __restrict__ wk, const _Float16* __restrict__ wv,
    const float* __restrict__ bq, const float* __restrict__ bk, const float* __restrict__ bv,
    _Float16* __restrict__ qh, _Float16* __restrict__ kh, _Float16* __restrict__ vt) {
  const int lane = threadIdx.x & 31;
  const int wave = threadIdx.x >> 5;
  const int half = lane >> 4;
  const int nl   = lane & 15;

  const int mt = blockIdx.x / (E_ / 64);
  const int nt = blockIdx.x % (E_ / 64);
  const int m0 = mt * 32 + (wave >> 2) * 16;
  const int n0 = nt * 64 + (wave & 3) * 16;

  const _Float16* xrow = xh + (size_t)(m0 + nl) * E_;
  const _Float16* qrow = wq + (size_t)(n0 + nl) * E_;
  const _Float16* krow = wk + (size_t)(n0 + nl) * E_;
  const _Float16* vrow = wv + (size_t)(n0 + nl) * E_;

  v8f cq = {}, ck = {}, cv = {};
  v16h a  = load_frag(xrow, half);
  v16h fq = load_frag(qrow, half);
  v16h fk = load_frag(krow, half);
  v16h fv = load_frag(vrow, half);
  for (int kc = 0; kc < E_ - 32; kc += 32) {
    v16h an  = load_frag(xrow + kc + 32, half);
    v16h fqn = load_frag(qrow + kc + 32, half);
    v16h fkn = load_frag(krow + kc + 32, half);
    v16h fvn = load_frag(vrow + kc + 32, half);
    cq = wmma_f16(a, fq, cq);
    ck = wmma_f16(a, fk, ck);
    cv = wmma_f16(a, fv, cv);
    a = an; fq = fqn; fk = fkn; fv = fvn;
  }
  cq = wmma_f16(a, fq, cq);
  ck = wmma_f16(a, fk, ck);
  cv = wmma_f16(a, fv, cv);

  const int bb = m0 / S_;
  const int s0 = m0 % S_;
  const int hh = n0 >> 6;
  const int d  = (n0 & 63) + nl;
  const float biasq = bq[n0 + nl];
  const float biask = bk[n0 + nl];
  const float biasv = bv[n0 + nl];

  _Float16* qbase = qh + ((size_t)((bb * H_ + hh) * S_) + s0 + half * 8) * DK_ + d;
  _Float16* kbase = kh + ((size_t)((bb * H_ + hh) * S_) + s0 + half * 8) * DK_ + d;
  #pragma unroll
  for (int v = 0; v < 8; ++v) {
    qbase[(size_t)v * DK_] = (_Float16)(cq[v] + biasq);
    kbase[(size_t)v * DK_] = (_Float16)(ck[v] + biask);
  }
  v8h vp;
  #pragma unroll
  for (int v = 0; v < 8; ++v) vp[v] = (_Float16)(cv[v] + biasv);
  *(v8h*)(vt + ((size_t)((bb * H_ + hh) * DK_) + d) * S_ + s0 + half * 8) = vp;
}

// ---------------------------------------------------------------- flash attention
// Block = 4 waves, one (b,h), 4 adjacent 16-row q-tiles. K/V key-chunks (32 keys)
// are staged into LDS once per block with async global->LDS copies, double-buffered.
__global__ __launch_bounds__(128) void attn(
    const _Float16* __restrict__ qh, const _Float16* __restrict__ kh,
    const _Float16* __restrict__ vt, const uint8_t* __restrict__ mask,
    _Float16* __restrict__ ctx) {
  __shared__ __align__(16) _Float16 kbuf[2][32 * DK_];   // [key][dk]
  __shared__ __align__(16) _Float16 vbuf[2][DK_ * 32];   // [dk][key]
  __shared__ __align__(16) _Float16 ptile_s[4][16 * 32];

  const int tid  = threadIdx.x;
  const int lane = tid & 31;
  const int wave = tid >> 5;
  const int half = lane >> 4;
  const int nl   = lane & 15;

  const int bh = blockIdx.x / (S_ / 64);
  const int qt = blockIdx.x % (S_ / 64);
  const int b  = bh / H_;
  const int h  = bh % H_;
  const int q0 = qt * 64 + wave * 16;

  const _Float16* Q     = qh + (size_t)(bh * S_ + q0) * DK_;
  const _Float16* Kbase = kh + (size_t)bh * S_ * DK_;
  const _Float16* Vbase = vt + (size_t)bh * DK_ * S_;
  const uint8_t*  mk    = mask + (size_t)b * S_ * S_;

  const v16h qa0 = load_frag(Q + (size_t)nl * DK_,      half);
  const v16h qa1 = load_frag(Q + (size_t)nl * DK_ + 32, half);

  v8f o0 = {}, o1 = {}, o2 = {}, o3 = {};
  float mrun[8], lrun[8];
  #pragma unroll
  for (int v = 0; v < 8; ++v) { mrun[v] = -__builtin_inff(); lrun[v] = 0.f; }

  _Float16* ptile = ptile_s[wave];

  // cooperative async stage of one 32-key chunk (K: 4KB contiguous, V: 64 rows x 64B)
  const int vrow0 = tid >> 2,        vcol0 = (tid & 3) * 8;          // granule tid
  const int vrow1 = (tid + 128) >> 2, vcol1 = ((tid + 128) & 3) * 8; // granule tid+128
  auto issue_chunk = [&](int kt, int buf) {
    async_ld16(Kbase + (size_t)kt * DK_ + tid * 8,          &kbuf[buf][tid * 8]);
    async_ld16(Kbase + (size_t)kt * DK_ + (tid + 128) * 8,  &kbuf[buf][(tid + 128) * 8]);
    async_ld16(Vbase + (size_t)vrow0 * S_ + kt + vcol0,     &vbuf[buf][tid * 8]);
    async_ld16(Vbase + (size_t)vrow1 * S_ + kt + vcol1,     &vbuf[buf][(tid + 128) * 8]);
  };

  issue_chunk(0, 0);
  for (int kt = 0, it = 0; kt < S_; kt += 32, ++it) {
    const int cur = it & 1;
    if (kt + 32 < S_) {
      issue_chunk(kt + 32, cur ^ 1);
      asm volatile("s_wait_asynccnt 0x4" ::: "memory");  // chunk 'cur' drained, next in flight
    } else {
      asm volatile("s_wait_asynccnt 0x0" ::: "memory");
    }
    __syncthreads();   // all waves' async copies for 'cur' are in LDS

    const _Float16* kb = kbuf[cur];
    const _Float16* vb = vbuf[cur];

    v8f s0 = {}, s1 = {};
    s0 = wmma_f16(qa0, load_frag(kb + nl * DK_,             half), s0);
    s0 = wmma_f16(qa1, load_frag(kb + nl * DK_ + 32,        half), s0);
    s1 = wmma_f16(qa0, load_frag(kb + (16 + nl) * DK_,      half), s1);
    s1 = wmma_f16(qa1, load_frag(kb + (16 + nl) * DK_ + 32, half), s1);

    #pragma unroll
    for (int v = 0; v < 8; ++v) {
      const int row = q0 + v + 8 * half;
      const uint8_t* mrow = mk + (size_t)row * S_;
      float f0 = mrow[kt + nl]      ? -1e9f : s0[v] * 0.125f;
      float f1 = mrow[kt + 16 + nl] ? -1e9f : s1[v] * 0.125f;

      float cur_m = fmaxf(f0, f1);
      cur_m = fmaxf(cur_m, __shfl_xor(cur_m, 1, 32));
      cur_m = fmaxf(cur_m, __shfl_xor(cur_m, 2, 32));
      cur_m = fmaxf(cur_m, __shfl_xor(cur_m, 4, 32));
      cur_m = fmaxf(cur_m, __shfl_xor(cur_m, 8, 32));
      float mnew  = fmaxf(mrun[v], cur_m);
      float alpha = __expf(mrun[v] - mnew);
      float p0 = __expf(f0 - mnew);
      float p1 = __expf(f1 - mnew);
      float ps = p0 + p1;
      ps += __shfl_xor(ps, 1, 32);
      ps += __shfl_xor(ps, 2, 32);
      ps += __shfl_xor(ps, 4, 32);
      ps += __shfl_xor(ps, 8, 32);
      lrun[v] = lrun[v] * alpha + ps;
      mrun[v] = mnew;
      o0[v] *= alpha; o1[v] *= alpha; o2[v] *= alpha; o3[v] *= alpha;
      ptile[(v + 8 * half) * 32 + nl]      = (_Float16)p0;
      ptile[(v + 8 * half) * 32 + 16 + nl] = (_Float16)p1;
    }
    asm volatile("s_wait_dscnt 0x0" ::: "memory");   // wave-internal cross-lane LDS RAW
    v16h pa = load_frag(ptile + nl * 32, half);

    o0 = wmma_f16(pa, load_frag(vb + (0 * 16 + nl) * 32, half), o0);
    o1 = wmma_f16(pa, load_frag(vb + (1 * 16 + nl) * 32, half), o1);
    o2 = wmma_f16(pa, load_frag(vb + (2 * 16 + nl) * 32, half), o2);
    o3 = wmma_f16(pa, load_frag(vb + (3 * 16 + nl) * 32, half), o3);

    __syncthreads();   // everyone done reading buf 'cur' before it is re-filled
  }

  _Float16* crow = ctx + ((size_t)b * S_ + q0 + 8 * half) * E_ + h * DK_ + nl;
  #pragma unroll
  for (int v = 0; v < 8; ++v) {
    float inv = 1.0f / lrun[v];
    _Float16* cr = crow + (size_t)v * E_;
    cr[0]  = (_Float16)(o0[v] * inv);
    cr[16] = (_Float16)(o1[v] * inv);
    cr[32] = (_Float16)(o2[v] * inv);
    cr[48] = (_Float16)(o3[v] * inv);
  }
}

// ---------------------------------------------------------------- output projection (f32 out)
__global__ __launch_bounds__(256) void out_proj(
    const _Float16* __restrict__ ctx, const _Float16* __restrict__ wo,
    const float* __restrict__ bo, float* __restrict__ out) {
  const int lane = threadIdx.x & 31;
  const int wave = threadIdx.x >> 5;
  const int half = lane >> 4;
  const int nl   = lane & 15;

  const int mt = blockIdx.x / (E_ / 64);
  const int nt = blockIdx.x % (E_ / 64);
  const int m0 = mt * 32 + (wave >> 2) * 16;
  const int n0 = nt * 64 + (wave & 3) * 16;

  const _Float16* arow = ctx + (size_t)(m0 + nl) * E_;
  const _Float16* brow = wo  + (size_t)(n0 + nl) * E_;

  v8f c = {};
  v16h a = load_frag(arow, half);
  v16h f = load_frag(brow, half);
  for (int kc = 0; kc < E_ - 32; kc += 32) {
    v16h an = load_frag(arow + kc + 32, half);
    v16h fn = load_frag(brow + kc + 32, half);
    c = wmma_f16(a, f, c);
    a = an; f = fn;
  }
  c = wmma_f16(a, f, c);

  const float bias = bo[n0 + nl];
  float* obase = out + (size_t)(m0 + 8 * half) * E_ + n0 + nl;
  #pragma unroll
  for (int v = 0; v < 8; ++v) obase[(size_t)v * E_] = c[v] + bias;
}

// ---------------------------------------------------------------- launcher
extern "C" void kernel_launch(void* const* d_in, const int* in_sizes, int n_in,
                              void* d_out, int out_size, void* d_ws, size_t ws_size,
                              hipStream_t stream) {
  (void)in_sizes; (void)n_in; (void)out_size;
  const float*   x    = (const float*)d_in[0];
  const uint8_t* mask = (const uint8_t*)d_in[1];   // jax bool -> 1 byte/elem
  const float* Wq = (const float*)d_in[2];
  const float* bq = (const float*)d_in[3];
  const float* Wk = (const float*)d_in[4];
  const float* bk = (const float*)d_in[5];
  const float* Wv = (const float*)d_in[6];
  const float* bv = (const float*)d_in[7];
  const float* Wo = (const float*)d_in[8];
  const float* bo = (const float*)d_in[9];

  const size_t MB = 1u << 20;
  if (ws_size < 48 * MB) return;  // layout below needs 48 MB
  char* ws = (char*)d_ws;
  _Float16* xh  = (_Float16*)(ws + 0 * MB);   // [M, E]        8 MB
  _Float16* wqh = (_Float16*)(ws + 8 * MB);   // [E, E]        2 MB
  _Float16* wkh = (_Float16*)(ws + 10 * MB);
  _Float16* wvh = (_Float16*)(ws + 12 * MB);
  _Float16* woh = (_Float16*)(ws + 14 * MB);
  _Float16* qh  = (_Float16*)(ws + 16 * MB);  // [B,H,S,DK]    8 MB
  _Float16* kh  = (_Float16*)(ws + 24 * MB);  // [B,H,S,DK]    8 MB
  _Float16* vt  = (_Float16*)(ws + 32 * MB);  // [B,H,DK,S]    8 MB
  _Float16* ctx = (_Float16*)(ws + 40 * MB);  // [M, E]        8 MB

  cvt_f32_f16<<<1024, 256, 0, stream>>>(x,  xh,  M_ * E_ / 4);
  cvt_f32_f16<<<1024, 256, 0, stream>>>(Wq, wqh, E_ * E_ / 4);
  cvt_f32_f16<<<1024, 256, 0, stream>>>(Wk, wkh, E_ * E_ / 4);
  cvt_f32_f16<<<1024, 256, 0, stream>>>(Wv, wvh, E_ * E_ / 4);
  cvt_f32_f16<<<1024, 256, 0, stream>>>(Wo, woh, E_ * E_ / 4);

  qkv_proj<<<(M_ / 32) * (E_ / 64), 256, 0, stream>>>(xh, wqh, wkh, wvh, bq, bk, bv,
                                                      qh, kh, vt);
  attn<<<B_ * H_ * (S_ / 64), 128, 0, stream>>>(qh, kh, vt, mask, ctx);
  out_proj<<<(M_ / 32) * (E_ / 64), 256, 0, stream>>>(ctx, woh, bo, (float*)d_out);
}
